// PaG_3633542332631
// MI455X (gfx1250) — matrix operand order repaired and checked by reading, not compiled, not run on hardware
//
#include <hip/hip_runtime.h>
#include <hip/hip_bf16.h>
#include <stdint.h>

// ---------------------------------------------------------------------------
// Problem constants (from reference)
// ---------------------------------------------------------------------------
#define S_LEN   512
#define D_DIM   300
#define DPAD    304      // D padded to multiple of 16 (N dim of GEMM)
#define P_DIM   32
#define NTAPS   11       // tap 0 = self, taps 1..10 = pair taps
#define CPT     10       // K-chunks of 32 per tap (320/32)
#define KCHUNKS (NTAPS*CPT)   // 110
#define REL     12
#define NB      5
#define NTHREADS 608     // 19 waves (wave32)

typedef __attribute__((ext_vector_type(16))) __bf16 v16bf;
typedef __attribute__((ext_vector_type(8)))  float  v8f;
typedef __attribute__((ext_vector_type(4)))  float  f4;

__device__ __forceinline__ uint32_t pack2bf(float a, float b) {
    // round-to-nearest-even f32 -> bf16, packed (a -> [15:0], b -> [31:16])
    uint32_t ua = __float_as_uint(a);
    uint32_t ub = __float_as_uint(b);
    ua = ua + 0x7FFFu + ((ua >> 16) & 1u);
    ub = ub + 0x7FFFu + ((ub >> 16) & 1u);
    return (ua >> 16) | (ub & 0xFFFF0000u);
}

// --- CDNA5 async global->LDS copy (ASYNCcnt-tracked, no VGPR round trip) ---
__device__ __forceinline__ void async_copy_b128(uint32_t lds_byte_addr,
                                                const uint32_t* gaddr) {
    asm volatile("global_load_async_to_lds_b128 %0, %1, off"
                 :: "v"(lds_byte_addr), "v"(gaddr)
                 : "memory");
}
__device__ __forceinline__ void wait_async0() {
#if __has_builtin(__builtin_amdgcn_s_wait_asynccnt)
    __builtin_amdgcn_s_wait_asynccnt(0);
#else
    asm volatile("s_wait_asynccnt 0x0" ::: "memory");
#endif
}

// ---------------------------------------------------------------------------
// K0: rel_emb_k / rel_emb_v broadcast fill.  268 MB of stores -> pure HBM BW.
// Non-temporal stores: the output exceeds L2 (192 MB); keep x/mtapP resident.
// ---------------------------------------------------------------------------
__global__ void rel_emb_fill(const float* __restrict__ pe_k,
                             const float* __restrict__ pe_v,
                             float* __restrict__ relk,
                             float* __restrict__ relv) {
    unsigned u = blockIdx.x * blockDim.x + threadIdx.x;   // < 512*512*8
    int j4 = u & 7;
    int j  = (u >> 3) & (S_LEN - 1);
    int i  = u >> 12;
    f4 vk = {0.f, 0.f, 0.f, 0.f};
    f4 vv = {0.f, 0.f, 0.f, 0.f};
    if (i >= j) {
        int pos = i - j + 1; if (pos > 512) pos = 512;
        vk = reinterpret_cast<const f4*>(pe_k + (size_t)pos * P_DIM)[j4];
        vv = reinterpret_cast<const f4*>(pe_v + (size_t)pos * P_DIM)[j4];
    }
    size_t base    = ((size_t)i * S_LEN + j) * (P_DIM / 4) + j4;
    size_t bstride = (size_t)S_LEN * S_LEN * (P_DIM / 4);
    f4* rk = reinterpret_cast<f4*>(relk);
    f4* rv = reinterpret_cast<f4*>(relv);
#pragma unroll
    for (int b = 0; b < 4; ++b) {
        __builtin_nontemporal_store(vk, &rk[base + (size_t)b * bstride]);
        __builtin_nontemporal_store(vv, &rv[base + (size_t)b * bstride]);
    }
}

// ---------------------------------------------------------------------------
// K1: total[b][d] = sum_i x[b,i,d]   (needed for the type-1 bucket constant)
// ---------------------------------------------------------------------------
__global__ void col_totals(const float* __restrict__ x,
                           float* __restrict__ totalws) {
    int b = blockIdx.x;
    int d = threadIdx.x;             // 0..319
    float acc = 0.f;
    if (d < D_DIM)
        for (int i = 0; i < S_LEN; ++i)
            acc += x[((size_t)b * S_LEN + i) * D_DIM + d];
    if (d < DPAD) totalws[b * DPAD + d] = (d < D_DIM) ? acc : 0.f;
}

// ---------------------------------------------------------------------------
// K2a: build tap matrices, PRE-PACKED in the WMMA B-operand pair layout,
// e-major so the GEMM B stage is a straight async memcpy:
//   mtapP[((t*10 + c)*DPAD + e)*16 + pr]  packs bf16(M_t[dd], M_t[dd+1]),
//   dd = c*32 + 2*pr,  zero-padded for dd>=300 or e>=300.
// Also writes W1 (f32) for the const kernel.
// ---------------------------------------------------------------------------
__global__ void build_mtap(const float* __restrict__ basis,
                           const float* __restrict__ comp,
                           const float* __restrict__ root,
                           uint32_t* __restrict__ mtapP,
                           float* __restrict__ w1ws) {
    int idx = blockIdx.x * blockDim.x + threadIdx.x;
    if (idx >= 160 * DPAD) return;
    int dp = idx / DPAD;
    int e  = idx - dp * DPAD;
    int d0 = 2 * dp;
    const float invS = 1.0f / (float)S_LEN;
    bool valid = (d0 < D_DIM) && (e < D_DIM);

    float w_0[REL], w_1[REL];
    if (valid) {
#pragma unroll
        for (int r = 0; r < REL; ++r) { w_0[r] = 0.f; w_1[r] = 0.f; }
#pragma unroll
        for (int p = 0; p < NB; ++p) {
            float b0 = basis[((size_t)p * D_DIM + d0) * D_DIM + e];
            float b1 = basis[((size_t)p * D_DIM + d0 + 1) * D_DIM + e];
#pragma unroll
            for (int r = 0; r < REL; ++r) {
                float cc = comp[r * NB + p];
                w_0[r] += cc * b0;
                w_1[r] += cc * b1;
            }
        }
        w1ws[(size_t)d0 * DPAD + e]       = w_0[1];
        w1ws[(size_t)(d0 + 1) * DPAD + e] = w_1[1];
    }
    int c  = d0 >> 5;
    int pr = (d0 & 31) >> 1;
    for (int t = 0; t < NTAPS; ++t) {
        float m0 = 0.f, m1 = 0.f;
        if (valid) {
            if (t == 0) {
                m0 = (w_0[0] - w_0[1]) * invS + root[(size_t)d0 * D_DIM + e];
                m1 = (w_1[0] - w_1[1]) * invS + root[(size_t)(d0 + 1) * D_DIM + e];
            } else {
                int r = REL - t;                     // W_{12-t}
                m0 = (w_0[r] - w_0[1]) * invS;
                m1 = (w_1[r] - w_1[1]) * invS;
            }
        }
        mtapP[(((size_t)t * CPT + c) * DPAD + e) * 16 + pr] = pack2bf(m0, m1);
    }
}

// ---------------------------------------------------------------------------
// K2b: const_b[e] = bias[e] + (total_b/S) @ W1
// ---------------------------------------------------------------------------
__global__ void build_const(const float* __restrict__ totalws,
                            const float* __restrict__ w1ws,
                            const float* __restrict__ bias,
                            float* __restrict__ constws) {
    int b = blockIdx.x;
    int e = threadIdx.x;             // 0..303
    float s = 0.f;
    if (e < D_DIM) {
        s = bias[e];
        const float invS = 1.0f / (float)S_LEN;
        for (int d = 0; d < D_DIM; ++d)
            s += totalws[b * DPAD + d] * invS * w1ws[(size_t)d * DPAD + e];
    }
    constws[b * DPAD + e] = s;
}

// ---------------------------------------------------------------------------
// K3: banded-conv GEMM with v_wmma_f32_16x16x32_bf16.
//   M = 2048 rows (b,j) : 128 blocks x 16 rows
//   N = 304 cols e      : 19 waves  x 16 cols
//   K = 110 chunks of 32 (11 taps x 320)
// Double-buffered LDS; B chunks streamed with global_load_async_to_lds_b128;
// A chunks built on the fly from x (bf16 pair-packed per ISA A layout).
// ---------------------------------------------------------------------------
__device__ __forceinline__ void stage_a_chunk(const float* __restrict__ x,
                                              int b, int j0, int tid, int kk,
                                              uint32_t* As) {
    if (tid < 256) {
        const int t  = kk / CPT;
        const int c  = kk % CPT;
        const int m  = tid >> 4;
        const int q  = tid & 15;
        const int dd = c * 32 + 2 * q;
        float a0 = 0.f, a1 = 0.f;
        if (dd < D_DIM) {
            if (t == 0) {
                const float* xr = x + (size_t)(b * S_LEN + j0 + m) * D_DIM;
                a0 = xr[dd]; a1 = xr[dd + 1];
            } else {
                int r1 = j0 + m + 2 * t - 1;
                if (r1 < S_LEN) {
                    const float* xr = x + (size_t)(b * S_LEN + r1) * D_DIM;
                    a0 += xr[dd]; a1 += xr[dd + 1];
                }
                if (r1 + 1 < S_LEN) {
                    const float* xr = x + (size_t)(b * S_LEN + r1 + 1) * D_DIM;
                    a0 += xr[dd]; a1 += xr[dd + 1];
                }
            }
        }
        As[m * 16 + q] = pack2bf(a0, a1);
    }
}

__global__ void __launch_bounds__(NTHREADS)
rgcn_gemm(const float* __restrict__ x,
          const uint32_t* __restrict__ mtapP,
          const float* __restrict__ constws,
          float* __restrict__ out) {
    __shared__ uint32_t As[2][16 * 16];        // [m][qpair]       2 x 1 KB
    __shared__ uint32_t Bs[2][DPAD * 16];      // [e][kpair]    2 x 19.5 KB

    const int tid  = threadIdx.x;
    const int lane = tid & 31;
    const int wave = tid >> 5;              // 0..18
    const int rb   = blockIdx.x;            // 0..127
    const int b    = rb >> 5;
    const int j0   = (rb & 31) << 4;
    const int e0   = wave << 4;
    const int half = lane >> 4;             // 0 | 1
    const int l16  = lane & 15;

    // 32-bit LDS byte addresses of the two B buffers (flat-low32 == LDS offset)
    uint32_t bs_lds[2];
    bs_lds[0] = (uint32_t)(uintptr_t)(&Bs[0][0]);
    bs_lds[1] = (uint32_t)(uintptr_t)(&Bs[1][0]);

    v8f acc = {};

    // ---- prologue: stage chunk 0 into buffer 0
    stage_a_chunk(x, b, j0, tid, 0, As[0]);
    {
        const uint32_t* src = mtapP;                  // chunk 0
        async_copy_b128(bs_lds[0] + (uint32_t)tid * 16u, src + (size_t)tid * 4);
        async_copy_b128(bs_lds[0] + (uint32_t)(tid + NTHREADS) * 16u,
                        src + (size_t)(tid + NTHREADS) * 4);
    }

    for (int kk = 0; kk < KCHUNKS; ++kk) {
        const int cur = kk & 1;
        const int nxt = cur ^ 1;

        wait_async0();        // my async copies of B(kk) have landed in LDS
        __syncthreads();      // everyone's copies + A(kk) stores are visible

        // ---- stage next chunk into the other buffer (overlaps with WMMA)
        if (kk + 1 < KCHUNKS) {
            const uint32_t* src = mtapP + (size_t)(kk + 1) * 16 * DPAD;
            async_copy_b128(bs_lds[nxt] + (uint32_t)tid * 16u,
                            src + (size_t)tid * 4);
            async_copy_b128(bs_lds[nxt] + (uint32_t)(tid + NTHREADS) * 16u,
                            src + (size_t)(tid + NTHREADS) * 4);
            stage_a_chunk(x, b, j0, tid, kk + 1, As[nxt]);
        }

        // ---- per-lane fragment loads (ISA 7.12.2 16-bit layouts)
        uint32_t av[8], bv[8];
        {
            const int m  = l16;
            const int qa = half * 4;        // A: q = (v>=4)*8 + half*4 + (v&3)
#pragma unroll
            for (int v = 0; v < 4; ++v) av[v]     = As[cur][m * 16 + qa + v];
#pragma unroll
            for (int v = 0; v < 4; ++v) av[4 + v] = As[cur][m * 16 + 8 + qa + v];
            const int pb = half * 4;        // B: pair = (v>=4)*8 + half*4 + (v&3)
            const uint32_t* brow = &Bs[cur][(e0 + l16) * 16 + pb];
#pragma unroll
            for (int v = 0; v < 8; ++v) bv[v] = brow[v + (v & 4)];
        }
        union { uint32_t u[8]; v16bf v; } A, B;
#pragma unroll
        for (int v = 0; v < 8; ++v) { A.u[v] = av[v]; B.u[v] = bv[v]; }

        acc = __builtin_amdgcn_wmma_f32_16x16x32_bf16(
                  /*neg_a=*/false, A.v, /*neg_b=*/false, B.v,
                  /*c_mod=*/(short)0, acc, /*reuse_a=*/false, /*reuse_b=*/false);
    }

    // ---- epilogue: D layout: VGPR v -> row v (lanes 0-15) / v+8 (lanes 16-31)
    const int col = e0 + l16;
    const float cb = constws[b * DPAD + col];
    if (col < D_DIM) {
#pragma unroll
        for (int v = 0; v < 8; ++v) {
            int m = v + half * 8;
            out[((size_t)(b * S_LEN + j0 + m)) * D_DIM + col] = acc[v] + cb;
        }
    }
}

// ---------------------------------------------------------------------------
// Host-side launch.  Workspace need: ~2.6 MB.
// ---------------------------------------------------------------------------
static inline size_t alignup(size_t v) { return (v + 255) & ~(size_t)255; }

extern "C" void kernel_launch(void* const* d_in, const int* in_sizes, int n_in,
                              void* d_out, int out_size, void* d_ws, size_t ws_size,
                              hipStream_t stream) {
    const float* x     = (const float*)d_in[0];
    // d_in[1] = adj_index: unused by the reference outputs
    const float* pe_k  = (const float*)d_in[2];
    const float* pe_v  = (const float*)d_in[3];
    const float* basis = (const float*)d_in[4];
    const float* comp  = (const float*)d_in[5];
    const float* root  = (const float*)d_in[6];
    const float* bias  = (const float*)d_in[7];

    // workspace carve
    char* ws = (char*)d_ws;
    float* totalws = (float*)ws;  ws += alignup(4 * DPAD * sizeof(float));
    float* w1ws    = (float*)ws;  ws += alignup((size_t)DPAD * DPAD * sizeof(float));
    float* constws = (float*)ws;  ws += alignup(4 * DPAD * sizeof(float));
    uint32_t* mtapP = (uint32_t*)ws;   // KCHUNKS*DPAD*16 dwords = 2.14 MB

    // output carve (tuple order: out, rel_emb_k, rel_emb_v)
    float* out  = (float*)d_out;
    float* relk = out  + (size_t)4 * S_LEN * D_DIM;
    float* relv = relk + (size_t)4 * S_LEN * S_LEN * P_DIM;

    // K0: 268 MB broadcast (dominant, bandwidth-bound)
    rel_emb_fill<<<8192, 256, 0, stream>>>(pe_k, pe_v, relk, relv);
    // K1: column totals
    col_totals<<<4, 320, 0, stream>>>(x, totalws);
    // K2a: tap matrices, WMMA-pair-packed bf16, e-major
    {
        int n = 160 * DPAD;
        build_mtap<<<(n + 255) / 256, 256, 0, stream>>>(basis, comp, root, mtapP, w1ws);
    }
    // K2b: per-batch constant row
    build_const<<<4, DPAD, 0, stream>>>(totalws, w1ws, bias, constws);
    // K3: WMMA banded-conv GEMM, async-LDS streamed B
    rgcn_gemm<<<128, NTHREADS, 0, stream>>>(x, mtapP, constws, out);
}
